// BaseSelfAttention_64879775973996
// MI455X (gfx1250) — matrix operand order repaired and checked by reading, not compiled, not run on hardware
//
#include <hip/hip_runtime.h>
#include <hip/hip_bf16.h>
#include <stdint.h>

#define DMODEL 2048
#define SEQ    2048
#define NHEAD  16
#define HDIM   128
#define NB     2

typedef _Float16 half16 __attribute__((ext_vector_type(16)));
typedef float    float8 __attribute__((ext_vector_type(8)));

static __device__ __forceinline__ float8 zero8() {
  float8 z;
  #pragma unroll
  for (int i = 0; i < 8; ++i) z[i] = 0.0f;
  return z;
}

static __device__ __forceinline__ float8 wmma_f16(half16 a, half16 b, float8 c) {
  // D = A(16x32 f16) * B(32x16 f16) + C(16x16 f32)
  return __builtin_amdgcn_wmma_f32_16x16x32_f16(false, a, false, b, (short)0, c,
                                                false, false);
}

// A-fragment 16x32 f16 from a row-major tile (leading dim `ld` in halves).
// ISA 7.12.2: lane M = lane%16, K-half = 8*(lane/16); element e -> K = e + 8*(e/8).
static __device__ __forceinline__ half16 load_a16x32(const _Float16* base, int ld,
                                                     int lane) {
  const int row = lane & 15, hi = lane >> 4;
  const unsigned* p =
      reinterpret_cast<const unsigned*>(base + (size_t)row * ld + hi * 8);
  union { half16 v; unsigned u[8]; } f;
  #pragma unroll
  for (int t = 0; t < 4; ++t) { f.u[t] = p[t]; f.u[4 + t] = p[8 + t]; }
  return f.v;
}

// B-fragment 32x16: lane holds column N = lane%16, K = 8*(lane/16) + e for
// e<8 and K = 16 + 8*(lane/16) + (e-8) for e>=8.  Source stored N-major: caller passes
// p = Bt + col*ld + k0.  Two 16-byte loads.
static __device__ __forceinline__ half16 load_b32x16(const _Float16* p, int hi) {
  const uint4* q = reinterpret_cast<const uint4*>(p + hi * 8);
  union { half16 v; uint4 u[2]; } f;
  f.u[0] = q[0]; f.u[1] = q[2];
  return f.v;
}

// wave32: row statistics live across the 16 lanes of a half-wave (C/D layout).
static __device__ __forceinline__ float redmax16(float x) {
  #pragma unroll
  for (int m = 1; m < 16; m <<= 1) x = fmaxf(x, __shfl_xor(x, m, 32));
  return x;
}
static __device__ __forceinline__ float redsum16(float x) {
  #pragma unroll
  for (int m = 1; m < 16; m <<= 1) x += __shfl_xor(x, m, 32);
  return x;
}

// --- CDNA5 async copy: global -> LDS, 16B per lane, tracked on ASYNCcnt. ----
// LDS byte address = low 32 bits of the generic shared pointer (ISA 10.2:
// LDS aperture uses addr[31:0]).
static __device__ __forceinline__ void async_g2l_b128(unsigned lds_byte_off,
                                                      const _Float16* g) {
  asm volatile("global_load_async_to_lds_b128 %0, %1, off"
               :: "v"(lds_byte_off), "v"(g)
               : "memory");
}
static __device__ __forceinline__ void wait_async0() {
  asm volatile("s_wait_asynccnt 0x0" ::: "memory");
}

// Stage a 16 x DMODEL f16 A-strip into LDS, one block cooperatively.
// 64 KiB total: 128 threads x 32 iters x 16 B.
static __device__ __forceinline__ void stage_a_strip(_Float16* As,
                                                     const _Float16* g,
                                                     int tid) {
  const unsigned lbase = (unsigned)(size_t)As;
  #pragma unroll 4
  for (int it = 0; it < 32; ++it) {
    const int idx = (it * 128 + tid) * 8;          // halves, 16 B chunks
    async_g2l_b128(lbase + (unsigned)idx * 2u, g + idx);
  }
  wait_async0();
  __syncthreads();
}

// ---------------------------------------------------------------------------
// W (f32, KxN row-major) -> Wt (f16, NxK row-major) for all four weights.
__global__ void k_transpose_w(const float* W0, const float* W1, const float* W2,
                              const float* W3, _Float16* T0, _Float16* T1,
                              _Float16* T2, _Float16* T3) {
  const float* W; _Float16* T;
  switch (blockIdx.z) {
    case 0: W = W0; T = T0; break;
    case 1: W = W1; T = T1; break;
    case 2: W = W2; T = T2; break;
    default: W = W3; T = T3; break;
  }
  __shared__ float tile[32][33];
  const int tx = threadIdx.x, ty = threadIdx.y;
  const int n0 = blockIdx.x * 32, k0 = blockIdx.y * 32;
  #pragma unroll
  for (int i = 0; i < 32; i += 8)
    tile[ty + i][tx] = W[(size_t)(k0 + ty + i) * DMODEL + n0 + tx];
  __syncthreads();
  #pragma unroll
  for (int i = 0; i < 32; i += 8)
    T[(size_t)(n0 + ty + i) * DMODEL + (k0 + tx)] = (_Float16)tile[tx][ty + i];
}

__global__ void k_cvt_f16(const float* s, _Float16* d, int n) {
  int i = blockIdx.x * blockDim.x + threadIdx.x;
  if (i < n) d[i] = (_Float16)s[i];
}

// ---------------------------------------------------------------------------
// QKV projection + bias + RoPE.  One wave -> 16 rows x 1 head (128 cols).
// The 4 waves of a block share one row-strip (4 consecutive heads), so the
// A-strip is async-staged into LDS once per block.
// mode (blockIdx.y): 0=Q (rope), 1=K (rope), 2=V (store transposed HDxS).
__global__ __launch_bounds__(128) void k_qkv_rope(
    const _Float16* __restrict__ xh, const _Float16* __restrict__ WtQ,
    const _Float16* __restrict__ WtK, const _Float16* __restrict__ WtV,
    const float* __restrict__ bq, const float* __restrict__ bk,
    const float* __restrict__ bv, _Float16* __restrict__ Qh,
    _Float16* __restrict__ Kh, _Float16* __restrict__ Vt) {
  __shared__ _Float16 As[16 * DMODEL];           // 64 KiB A-strip
  const int lane = threadIdx.x & 31, wv = threadIdx.x >> 5;
  const int wave = blockIdx.x * 4 + wv;          // 0..4095
  const int strip = wave >> 4, head = wave & 15; // 256 row-strips x 16 heads
  const int mode = blockIdx.y;
  const _Float16* Wt = (mode == 0) ? WtQ : (mode == 1) ? WtK : WtV;
  const float* bias  = (mode == 0) ? bq  : (mode == 1) ? bk  : bv;
  const int ln = lane & 15, hi = lane >> 4;
  const int row0 = strip * 16;                   // identical for all 4 waves

  stage_a_strip(As, xh + (size_t)row0 * DMODEL, threadIdx.x);

  float8 acc[8];
  #pragma unroll
  for (int n = 0; n < 8; ++n) acc[n] = zero8();

  for (int k0 = 0; k0 < DMODEL; k0 += 32) {
    half16 a = load_a16x32(As + k0, DMODEL, lane);   // LDS
    #pragma unroll
    for (int n = 0; n < 8; ++n) {
      const _Float16* bp =
          Wt + (size_t)(head * HDIM + n * 16 + ln) * DMODEL + k0;
      acc[n] = wmma_f16(a, load_b32x16(bp, hi), acc[n]);
    }
  }
  #pragma unroll
  for (int n = 0; n < 8; ++n) {
    float bb = bias[head * HDIM + n * 16 + ln];
    #pragma unroll
    for (int v = 0; v < 8; ++v) acc[n][v] += bb;
  }

  const int sbase = row0 & (SEQ - 1);
  const int bidx = row0 >> 11;           // row0 / SEQ
  const int bh = bidx * NHEAD + head;

  if (mode < 2) {
    // RoPE: pair (i, i+64) -> fragments n and n+4, same lane/slot.
    #pragma unroll
    for (int n = 0; n < 4; ++n) {
      float j = (float)(n * 16 + ln);
      float inv = __powf(10000.0f, -(2.0f * j) * (1.0f / 128.0f));
      #pragma unroll
      for (int v = 0; v < 8; ++v) {
        float pos = (float)(sbase + v + 8 * hi);
        float ang = pos * inv;
        float c = __cosf(ang), s = __sinf(ang);
        float t1 = acc[n][v], t2 = acc[n + 4][v];
        acc[n][v]     = t1 * c - t2 * s;
        acc[n + 4][v] = t2 * c + t1 * s;
      }
    }
    _Float16* dst = ((mode == 0) ? Qh : Kh) + (size_t)bh * SEQ * HDIM;
    #pragma unroll
    for (int n = 0; n < 8; ++n)
      #pragma unroll
      for (int v = 0; v < 8; ++v) {
        int s = sbase + v + 8 * hi;
        dst[(size_t)s * HDIM + n * 16 + ln] = (_Float16)acc[n][v];
      }
  } else {
    _Float16* dst = Vt + (size_t)bh * HDIM * SEQ;   // V^T: [hd][s]
    #pragma unroll
    for (int n = 0; n < 8; ++n)
      #pragma unroll
      for (int v = 0; v < 8; ++v) {
        int s = sbase + v + 8 * hi;
        dst[(size_t)(n * 16 + ln) * SEQ + s] = (_Float16)acc[n][v];
      }
  }
}

// ---------------------------------------------------------------------------
// Flash attention: one wave owns a 16-row q-strip, loops causally over 32-wide
// key tiles.  Per iter: 8 WMMA (scores) + online softmax + 8 WMMA (P*V).
// No block barriers: trip counts diverge per wave (causal), K/V stream from L2.
__global__ __launch_bounds__(128) void k_flash_attn(
    const _Float16* __restrict__ Qh, const _Float16* __restrict__ Kh,
    const _Float16* __restrict__ Vt, _Float16* __restrict__ Oh) {
  __shared__ _Float16 plds[4 * 16 * 32];   // per-wave 16x32 P staging
  const int lane = threadIdx.x & 31, wv = threadIdx.x >> 5;
  const int bh = blockIdx.x;
  const int q0 = blockIdx.y * 64 + wv * 16;
  const int ln = lane & 15, hi = lane >> 4;
  const _Float16* Qb = Qh + (size_t)bh * SEQ * HDIM;
  const _Float16* Kb = Kh + (size_t)bh * SEQ * HDIM;
  const _Float16* Vb = Vt + (size_t)bh * HDIM * SEQ;

  half16 qf[4];
  #pragma unroll
  for (int i = 0; i < 4; ++i)
    qf[i] = load_a16x32(Qb + (size_t)q0 * HDIM + i * 32, HDIM, lane);

  float8 oacc[8];
  #pragma unroll
  for (int j = 0; j < 8; ++j) oacc[j] = zero8();
  float mrun[8], lrun[8];
  #pragma unroll
  for (int v = 0; v < 8; ++v) { mrun[v] = -1e30f; lrun[v] = 0.0f; }

  const float scale = 0.08838834764831845f;  // 1/sqrt(128)
  _Float16* P = plds + wv * 512;

  for (int kt = 0; kt < q0 + 16; kt += 32) {     // causal: keys <= q only
    float8 s0 = zero8(), s1 = zero8();
    #pragma unroll
    for (int i = 0; i < 4; ++i) {
      s0 = wmma_f16(qf[i], load_b32x16(Kb + (size_t)(kt + ln) * HDIM + i * 32, hi), s0);
      s1 = wmma_f16(qf[i], load_b32x16(Kb + (size_t)(kt + 16 + ln) * HDIM + i * 32, hi), s1);
    }
    #pragma unroll
    for (int v = 0; v < 8; ++v) {
      int row = q0 + v + 8 * hi;
      float a0 = (kt + ln      <= row) ? s0[v] * scale : -1e30f;
      float a1 = (kt + 16 + ln <= row) ? s1[v] * scale : -1e30f;
      float mnew = fmaxf(mrun[v], redmax16(fmaxf(a0, a1)));
      float r = __expf(mrun[v] - mnew);
      mrun[v] = mnew;
      float p0 = __expf(a0 - mnew);
      float p1 = __expf(a1 - mnew);
      lrun[v] = lrun[v] * r + redsum16(p0 + p1);
      #pragma unroll
      for (int j = 0; j < 8; ++j) oacc[j][v] *= r;
      int prow = v + 8 * hi;
      P[prow * 32 + ln]      = (_Float16)p0;
      P[prow * 32 + 16 + ln] = (_Float16)p1;
    }
    // D-fragment -> A-fragment transpose via per-wave LDS; CDNA5 split counter.
    asm volatile("s_wait_dscnt 0" ::: "memory");
    half16 pf = load_a16x32(P, 32, lane);
    #pragma unroll
    for (int j = 0; j < 8; ++j)
      oacc[j] = wmma_f16(pf,
          load_b32x16(Vb + (size_t)(j * 16 + ln) * SEQ + kt, hi), oacc[j]);
  }

  const int bb = bh >> 4, h = bh & 15;
  #pragma unroll
  for (int j = 0; j < 8; ++j)
    #pragma unroll
    for (int v = 0; v < 8; ++v) {
      int s = q0 + v + 8 * hi;
      float o = oacc[j][v] / lrun[v];
      Oh[(size_t)(bb * SEQ + s) * DMODEL + h * HDIM + j * 16 + ln] = (_Float16)o;
    }
}

// ---------------------------------------------------------------------------
// out = attn @ Wo + bo  (f32 output).  Same block-shared async A staging.
__global__ __launch_bounds__(128) void k_out_proj(
    const _Float16* __restrict__ Oh, const _Float16* __restrict__ WtO,
    const float* __restrict__ bo, float* __restrict__ out) {
  __shared__ _Float16 As[16 * DMODEL];           // 64 KiB A-strip
  const int lane = threadIdx.x & 31, wv = threadIdx.x >> 5;
  const int wave = blockIdx.x * 4 + wv;
  const int strip = wave >> 4, ct = wave & 15;
  const int ln = lane & 15, hi = lane >> 4;
  const int row0 = strip * 16;

  stage_a_strip(As, Oh + (size_t)row0 * DMODEL, threadIdx.x);

  float8 acc[8];
  #pragma unroll
  for (int n = 0; n < 8; ++n) acc[n] = zero8();

  for (int k0 = 0; k0 < DMODEL; k0 += 32) {
    half16 a = load_a16x32(As + k0, DMODEL, lane);   // LDS
    #pragma unroll
    for (int n = 0; n < 8; ++n) {
      const _Float16* bp =
          WtO + (size_t)(ct * HDIM + n * 16 + ln) * DMODEL + k0;
      acc[n] = wmma_f16(a, load_b32x16(bp, hi), acc[n]);
    }
  }
  #pragma unroll
  for (int n = 0; n < 8; ++n) {
    int col = ct * HDIM + n * 16 + ln;
    float bb = bo[col];
    #pragma unroll
    for (int v = 0; v < 8; ++v)
      out[(size_t)(row0 + v + 8 * hi) * DMODEL + col] = acc[n][v] + bb;
  }
}

// ---------------------------------------------------------------------------
extern "C" void kernel_launch(void* const* d_in, const int* in_sizes, int n_in,
                              void* d_out, int out_size, void* d_ws,
                              size_t ws_size, hipStream_t stream) {
  (void)in_sizes; (void)n_in; (void)out_size; (void)ws_size;
  const float* x  = (const float*)d_in[0];
  // d_in[1] = attention_mask: causal, implemented analytically
  const float* Wq = (const float*)d_in[2];
  const float* bq = (const float*)d_in[3];
  const float* Wk = (const float*)d_in[4];
  const float* bk = (const float*)d_in[5];
  const float* Wv = (const float*)d_in[6];
  const float* bv = (const float*)d_in[7];
  const float* Wo = (const float*)d_in[8];
  const float* bo = (const float*)d_in[9];
  float* out = (float*)d_out;

  char* ws = (char*)d_ws;
  const size_t ACT = (size_t)NB * SEQ * DMODEL * sizeof(_Float16);  // 16 MiB
  const size_t WT  = (size_t)DMODEL * DMODEL * sizeof(_Float16);    //  8 MiB
  _Float16* xh  = (_Float16*)(ws);                 // reused as attn output Oh
  _Float16* WtQ = (_Float16*)(ws + ACT);
  _Float16* WtK = (_Float16*)(ws + ACT + 1 * WT);
  _Float16* WtV = (_Float16*)(ws + ACT + 2 * WT);
  _Float16* WtO = (_Float16*)(ws + ACT + 3 * WT);
  _Float16* Qh  = (_Float16*)(ws + ACT + 4 * WT);
  _Float16* Kh  = (_Float16*)(ws + ACT + 4 * WT + 1 * ACT);
  _Float16* Vt  = (_Float16*)(ws + ACT + 4 * WT + 2 * ACT);
  // total workspace: 4*16 MiB + 4*8 MiB = 96 MiB

  k_transpose_w<<<dim3(DMODEL / 32, DMODEL / 32, 4), dim3(32, 8), 0, stream>>>(
      Wq, Wk, Wv, Wo, WtQ, WtK, WtV, WtO);

  const int nx = NB * SEQ * DMODEL;
  k_cvt_f16<<<(nx + 255) / 256, 256, 0, stream>>>(x, xh, nx);

  // 4096 waves per mode: 256 row-strips x 16 heads; 3 modes (Q,K,V)
  k_qkv_rope<<<dim3(1024, 3), 128, 0, stream>>>(xh, WtQ, WtK, WtV, bq, bk, bv,
                                                Qh, Kh, Vt);

  // 32 (b,h) x 32 q-tiles of 64 rows; xh no longer needed -> reuse as Oh
  k_flash_attn<<<dim3(NB * NHEAD, SEQ / 64), 128, 0, stream>>>(Qh, Kh, Vt, xh);

  k_out_proj<<<dim3(1024), 128, 0, stream>>>(xh, WtO, bo, out);
}
